// StaticCompositeLinear_16716012716090
// MI455X (gfx1250) — matrix operand order repaired and checked by reading, not compiled, not run on hardware
//
#include <hip/hip_runtime.h>
#include <stdint.h>

#define T_TOKENS 16384   // B*S = 4*4096
#define DDIM     512
#define NEXP     8

typedef __attribute__((ext_vector_type(16))) __bf16 v16bf;
typedef __attribute__((ext_vector_type(8)))  __bf16 v8bf;
typedef __attribute__((ext_vector_type(8)))  float  v8f;
typedef __attribute__((address_space(3)))    uint16_t lds_u16;

__device__ __forceinline__ uint16_t f32_to_bf16_rne(float f) {
    union { float f; uint32_t u; } cv; cv.f = f;
    uint32_t u = cv.u;
    uint32_t r = u + 0x7FFFu + ((u >> 16) & 1u);   // round-to-nearest-even
    return (uint16_t)(r >> 16);
}

// ---------------- f32 -> bf16 conversion (one-time prep) ----------------
__global__ __launch_bounds__(256) void cvt_bf16_kernel(const float* __restrict__ src,
                                                       uint16_t* __restrict__ dst, int n) {
    int i = blockIdx.x * blockDim.x + threadIdx.x;
    int stride = gridDim.x * blockDim.x;
    for (; i < n; i += stride) dst[i] = f32_to_bf16_rne(src[i]);
}

// ---------------- gating: logits -> softmax coeffs [token][8] ----------------
__global__ __launch_bounds__(256) void gating_kernel(const float* __restrict__ x,
                                                     const float* __restrict__ mw,
                                                     const float* __restrict__ mb,
                                                     float* __restrict__ coeffs) {
    int token = blockIdx.x * 8 + (threadIdx.x >> 5);
    int lane  = threadIdx.x & 31;
    const float* xr = x + (size_t)token * DDIM;

    float part[NEXP];
#pragma unroll
    for (int e = 0; e < NEXP; ++e) part[e] = 0.0f;

    for (int t = 0; t < DDIM / 32; ++t) {
        float xv = xr[lane + 32 * t];
#pragma unroll
        for (int e = 0; e < NEXP; ++e)
            part[e] += xv * mw[e * DDIM + lane + 32 * t];
    }
#pragma unroll
    for (int e = 0; e < NEXP; ++e) {
#pragma unroll
        for (int off = 16; off > 0; off >>= 1)
            part[e] += __shfl_xor(part[e], off, 32);
    }
    if (lane == 0) {
        float lg[NEXP];
        float mx = -1e30f;
#pragma unroll
        for (int e = 0; e < NEXP; ++e) { lg[e] = part[e] + mb[e]; mx = fmaxf(mx, lg[e]); }
        float sum = 0.0f;
#pragma unroll
        for (int e = 0; e < NEXP; ++e) { lg[e] = __expf(lg[e] - mx); sum += lg[e]; }
        float inv = 1.0f / sum;
#pragma unroll
        for (int e = 0; e < NEXP; ++e) coeffs[(size_t)token * NEXP + e] = lg[e] * inv;
    }
}

// ---------------- main MoE kernel ----------------
// Block = 8 waves sharing one 16-wide N-tile; wave w owns TWO M-tiles (32 rows).
// B tile (8 experts x 16 n x 512 k bf16 = 128KB) staged in LDS in FRAGMENT-READY
// order via per-lane global_load_async_to_lds_b128 scatter:
//   region(kblock,e) = 1KB at (kblock*8+e)*1024:
//     [ 0..511]: lane L's low 16B  (W[e][n0+(L&15)][k0+(L>>4)*8 + 0..7])
//     [512..1023]: lane L's high 16B (                       ... + 16..23])
// -> in the k-loop all 16 ds_load_b128 use one base register + small immediates,
//    and each half-wave's 16 lanes hit all 64 LDS banks (conflict-free).
__global__ __launch_bounds__(256) void moe_wmma_kernel(
        const uint16_t* __restrict__ xbf,   // [T][512] bf16
        const uint16_t* __restrict__ wbf,   // [8][512][512] bf16 (o-major, i-contig)
        const float*    __restrict__ coeffs,// [T][8]
        const float*    __restrict__ biases,// [8][512]
        float*          __restrict__ y) {   // [T][512]
    extern __shared__ uint16_t ldsb[];      // 65536 elems = 131,072 B

    const int tid  = threadIdx.x;
    const int wave = tid >> 5;
    const int lane = tid & 31;
    const int half = lane >> 4;     // 0: lanes 0-15, 1: lanes 16-31
    const int l15  = lane & 15;
    const int n0   = blockIdx.y * 16;
    const int m0   = blockIdx.x * 256 + wave * 32;   // 32 rows per wave

    // ---- cooperative async scatter of B tile into fragment-ready LDS ----
    // chunk c = (row = e*16+n) * 64 + kc ; global 16B chunks are consecutive per
    // lane (coalesced); LDS destination is the swizzled fragment slot.
    lds_u16* lb3 = (lds_u16*)ldsb;
#pragma unroll 4
    for (int i = 0; i < 32; ++i) {
        int c   = tid + 256 * i;        // 0..8191
        int row = c >> 6;               // 0..127 (= e*16 + n)
        int kc  = c & 63;               // 16B chunk within the 1KB W row
        int e = row >> 4, n = row & 15;
        int kblock = kc >> 2;           // k0/32
        int hsel   = kc & 1;            // which half-wave owns this chunk
        int h      = (kc >> 1) & 1;     // blo/bhi
        int L      = hsel * 16 + n;     // destination lane slot
        const uint16_t* g = wbf + ((size_t)e * DDIM + (n0 + n)) * DDIM + kc * 8;
        lds_u16* l = lb3 + (size_t)(((kblock * 8 + e) * 2 + h) * 32 + L) * 8;
        asm volatile("global_load_async_to_lds_b128 %0, %1, off"
                     :: "v"(l), "v"(g) : "memory");
    }
    asm volatile("s_wait_asynccnt 0x0" ::: "memory");
    __syncthreads();

    // ---- k-loop: A from global, B from LDS, 2 M-tiles x 8 expert accumulators ----
    v8f acc0[NEXP] = {};
    v8f acc1[NEXP] = {};
    const __bf16* arow0 = (const __bf16*)xbf + (size_t)(m0 + l15) * DDIM;
    const __bf16* arow1 = (const __bf16*)xbf + (size_t)(m0 + 16 + l15) * DDIM;
    const __bf16* bframe = (const __bf16*)ldsb + lane * 8;   // lane's 16B slot

    for (int k0 = 0; k0 < DDIM; k0 += 32) {
        // A (16-bit 16x32): lanes 0-15 K={0..7,16..23}, lanes 16-31 K={8..15,24..31}
        v8bf a0lo = *(const v8bf*)(arow0 + k0 + half * 8);
        v8bf a0hi = *(const v8bf*)(arow0 + k0 + 16 + half * 8);
        v8bf a1lo = *(const v8bf*)(arow1 + k0 + half * 8);
        v8bf a1hi = *(const v8bf*)(arow1 + k0 + 16 + half * 8);
        v16bf a0 = __builtin_shufflevector(a0lo, a0hi,
                     0,1,2,3,4,5,6,7,8,9,10,11,12,13,14,15);
        v16bf a1 = __builtin_shufflevector(a1lo, a1hi,
                     0,1,2,3,4,5,6,7,8,9,10,11,12,13,14,15);
        const __bf16* bbase = bframe + (k0 >> 5) * 4096;  // region row: 8KB/kblock
#pragma unroll
        for (int e = 0; e < NEXP; ++e) {
            v8bf blo = *(const v8bf*)(bbase + e * 512);        // imm = e*1024 B
            v8bf bhi = *(const v8bf*)(bbase + e * 512 + 256);  // imm = e*1024+512 B
            v16bf b = __builtin_shufflevector(blo, bhi,
                        0,1,2,3,4,5,6,7,8,9,10,11,12,13,14,15);
            acc0[e] = __builtin_amdgcn_wmma_f32_16x16x32_bf16(
                          false, a0, false, b, (short)0, acc0[e], false, false);
            acc1[e] = __builtin_amdgcn_wmma_f32_16x16x32_bf16(
                          false, a1, false, b, (short)0, acc1[e], false, false);
        }
    }

    // ---- epilogue: y[row][n] = sum_e coeff[row][e] * (acc_e + bias[e][n]) ----
    float be[NEXP];
#pragma unroll
    for (int e = 0; e < NEXP; ++e) be[e] = biases[e * DDIM + n0 + l15];

#pragma unroll
    for (int j = 0; j < 8; ++j) {
        int row = m0 + half * 8 + j;            // C/D layout: VGPR j -> M = j + 8*half
        const float* cf = coeffs + (size_t)row * NEXP;
        float s = 0.0f;
#pragma unroll
        for (int e = 0; e < NEXP; ++e) s += cf[e] * (acc0[e][j] + be[e]);
        y[(size_t)row * DDIM + n0 + l15] = s;
    }
#pragma unroll
    for (int j = 0; j < 8; ++j) {
        int row = m0 + 16 + half * 8 + j;
        const float* cf = coeffs + (size_t)row * NEXP;
        float s = 0.0f;
#pragma unroll
        for (int e = 0; e < NEXP; ++e) s += cf[e] * (acc1[e][j] + be[e]);
        y[(size_t)row * DDIM + n0 + l15] = s;
    }
}

extern "C" void kernel_launch(void* const* d_in, const int* in_sizes, int n_in,
                              void* d_out, int out_size, void* d_ws, size_t ws_size,
                              hipStream_t stream) {
    const float* x  = (const float*)d_in[0];   // [4,4096,512]
    const float* ew = (const float*)d_in[1];   // [8,512,512]
    const float* eb = (const float*)d_in[2];   // [8,512]
    const float* mw = (const float*)d_in[3];   // [8,512]
    const float* mb = (const float*)d_in[4];   // [8]
    float* y = (float*)d_out;

    uint8_t* ws = (uint8_t*)d_ws;
    uint16_t* xbf    = (uint16_t*)ws;                                    // 16 MB
    uint16_t* wbf    = (uint16_t*)(ws + (size_t)T_TOKENS * DDIM * 2);    // 4 MB
    float*    coeffs = (float*)(ws + (size_t)T_TOKENS * DDIM * 2
                                   + (size_t)NEXP * DDIM * DDIM * 2);    // 512 KB

    cvt_bf16_kernel<<<2048, 256, 0, stream>>>(x,  xbf, T_TOKENS * DDIM);
    cvt_bf16_kernel<<<1024, 256, 0, stream>>>(ew, wbf, NEXP * DDIM * DDIM);
    gating_kernel<<<T_TOKENS / 8, 256, 0, stream>>>(x, mw, mb, coeffs);

    dim3 grid(T_TOKENS / 256, DDIM / 16);      // 64 M-supertiles x 32 N-tiles
    size_t ldsbytes = (size_t)NEXP * 16 * 1024;   // 131,072 B (16 kblocks x 8 experts x 1KB)
    moe_wmma_kernel<<<grid, 256, ldsbytes, stream>>>(xbf, wbf, coeffs, eb, y);
}
